// AFEModule_46067819217433
// MI455X (gfx1250) — matrix operand order repaired and checked by reading, not compiled, not run on hardware
//
#include <hip/hip_runtime.h>
#include <hip/hip_bf16.h>

// ---------------------------------------------------------------------------
// AFE pipeline for MI455X (gfx1250, wave32):
//   fft2(+shift) -> center-mag MLP -> aniso 7x7 kernel -> conv (bf16 out) ->
//   WMMA channel mix (bf16 A/B, f32 acc, 64-pixel tiles) -> ifft2(+shift)+x.
// ---------------------------------------------------------------------------

#define HH     128
#define CCH    256
#define HWNUM  16384
#define LSTR   130            // padded LDS row stride for FFT
#define XSTR   40             // padded LDS row stride (bf16) for GEMM staging

typedef __attribute__((ext_vector_type(16))) __bf16 v16bf;
typedef __attribute__((ext_vector_type(8)))  float  v8f;

__device__ __forceinline__ int bitrev7(int v) {
  return (int)(__brev((unsigned)v) >> 25);
}

__device__ __forceinline__ unsigned short f2bf(float f) {
  unsigned int u = __float_as_uint(f);
  unsigned int r = (u + 0x7FFFu + ((u >> 16) & 1u)) >> 16;   // RNE
  return (unsigned short)r;
}

// ------------------------------ FFT helpers --------------------------------

__device__ __forceinline__ void afe_fft_rows(float* re, float* im, float sgn) {
  for (int s = 0; s < 7; ++s) {
    int half = 1 << s;
    float a0 = sgn * 3.14159265358979323846f / (float)half;
    for (int idx = threadIdx.x; idx < HH * 64; idx += blockDim.x) {
      int r   = idx >> 6;
      int j   = idx & 63;
      int pos = j & (half - 1);
      int i0  = r * LSTR + (((j >> s) << (s + 1)) + pos);
      int i1  = i0 + half;
      float ang = a0 * (float)pos;
      float wc = __cosf(ang), ws = __sinf(ang);
      float xr = re[i1], xi = im[i1];
      float tr = xr * wc - xi * ws;
      float ti = xr * ws + xi * wc;
      float ur = re[i0], ui = im[i0];
      re[i0] = ur + tr; im[i0] = ui + ti;
      re[i1] = ur - tr; im[i1] = ui - ti;
    }
    __syncthreads();
  }
}

__device__ __forceinline__ void afe_fft_cols(float* re, float* im, float sgn) {
  for (int s = 0; s < 7; ++s) {
    int half = 1 << s;
    float a0 = sgn * 3.14159265358979323846f / (float)half;
    for (int idx = threadIdx.x; idx < HH * 64; idx += blockDim.x) {
      int c   = idx & 127;
      int j   = idx >> 7;
      int pos = j & (half - 1);
      int r0  = ((j >> s) << (s + 1)) + pos;
      int i0  = r0 * LSTR + c;
      int i1  = i0 + half * LSTR;
      float ang = a0 * (float)pos;
      float wc = __cosf(ang), ws = __sinf(ang);
      float xr = re[i1], xi = im[i1];
      float tr = xr * wc - xi * ws;
      float ti = xr * ws + xi * wc;
      float ur = re[i0], ui = im[i0];
      re[i0] = ur + tr; im[i0] = ui + ti;
      re[i1] = ur - tr; im[i1] = ui - ti;
    }
    __syncthreads();
  }
}

__device__ __forceinline__ void afe_swap_rows_bitrev(float* re, float* im) {
  for (int idx = threadIdx.x; idx < HH * HH; idx += blockDim.x) {
    int r = idx >> 7, c = idx & 127;
    int br = bitrev7(r);
    if (r < br) {
      int a = r * LSTR + c, b = br * LSTR + c;
      float t = re[a]; re[a] = re[b]; re[b] = t;
      t = im[a]; im[a] = im[b]; im[b] = t;
    }
  }
  __syncthreads();
}

// ------------------- forward FFT + shifted-center magnitude ----------------
// fftshift folded in: FFT((-1)^(r+c) * x) == fftshift(FFT(x)) for even N.

__global__ void afe_fft2_fwd(const float* __restrict__ x,
                             float* __restrict__ xfr, float* __restrict__ xfi,
                             float* __restrict__ mag49) {
  extern __shared__ float lds[];
  float* re = lds;
  float* im = lds + HH * LSTR;
  int img = blockIdx.x;            // b*C + c
  int b   = img >> 8;
  const float* src = x + (size_t)img * HWNUM;

  for (int i = threadIdx.x; i < HWNUM; i += blockDim.x) {
    int r = i >> 7, c = i & 127;
    float v = src[i];
    if ((r + c) & 1) v = -v;
    int bc = bitrev7(c);
    re[r * LSTR + bc] = v;
    im[r * LSTR + bc] = 0.f;
  }
  __syncthreads();
  afe_fft_rows(re, im, -1.f);
  afe_swap_rows_bitrev(re, im);
  afe_fft_cols(re, im, -1.f);

  const float scale = 0.0078125f;  // 1/128 (ortho for 128x128)
  float* dr = xfr + (size_t)img * HWNUM;
  float* di = xfi + (size_t)img * HWNUM;
  for (int i = threadIdx.x; i < HWNUM; i += blockDim.x) {
    int r = i >> 7, c = i & 127;
    float vr = re[r * LSTR + c] * scale;
    float vi = im[r * LSTR + c] * scale;
    dr[i] = vr; di[i] = vi;
    if (r >= 61 && r <= 67 && c >= 61 && c <= 67) {
      float m = sqrtf(vr * vr + vi * vi) * (1.0f / 256.0f);
      atomicAdd(&mag49[b * 64 + (r - 61) * 7 + (c - 61)], m);
    }
  }
}

// --------------------------- MLP + aniso kernel ----------------------------

__global__ void afe_mlp_kern(const float* __restrict__ mag49,
                             const float* __restrict__ w1, const float* __restrict__ b1,
                             const float* __restrict__ w2, const float* __restrict__ b2,
                             float* __restrict__ kern) {
  int b = blockIdx.x;
  __shared__ float c49[49];
  __shared__ float hid[32];
  __shared__ float prm[3];
  __shared__ float kv[49];
  __shared__ float kinv;
  int t = threadIdx.x;             // 64 threads
  if (t < 49) c49[t] = mag49[b * 64 + t];
  __syncthreads();
  if (t < 32) {
    float acc = b1[t];
    for (int i = 0; i < 49; ++i) acc = fmaf(c49[i], w1[i * 32 + t], acc);
    hid[t] = fmaxf(acc, 0.f);
  }
  __syncthreads();
  if (t < 3) {
    float acc = b2[t];
    for (int j = 0; j < 32; ++j) acc = fmaf(hid[j], w2[j * 3 + t], acc);
    prm[t] = acc;
  }
  __syncthreads();
  float theta = atan2f(prm[0], prm[1]) * 0.5f + 1.57079632679f;
  float lam1  = expf(prm[2]);
  float lam2  = 1.0f / (lam1 + 1e-8f);
  float ct = cosf(theta), st = sinf(theta);
  if (t < 49) {
    int r = t / 7, c = t % 7;
    float y  = (float)(r - 3), xx = (float)(c - 3);
    float xr = xx * ct + y * st;
    float yr = -xx * st + y * ct;
    float e  = expf(-(xr * xr / (2.f * lam1 * lam1) + yr * yr / (2.f * lam2 * lam2)));
    kv[t] = e;
  }
  __syncthreads();
  if (t == 0) {
    float s = 0.f;
    for (int i = 0; i < 49; ++i) s += kv[i];
    kinv = 1.0f / (s + 1e-8f);
  }
  __syncthreads();
  if (t < 49) kern[b * 64 + t] = kv[t] * kinv;
}

// -------------------- 7x7 conv, f32 in -> bf16 out -------------------------

__global__ void afe_conv7(const float* __restrict__ in, unsigned short* __restrict__ outp,
                          const float* __restrict__ kern) {
  int img = blockIdx.z;            // b*C + c
  int b   = img >> 8;
  __shared__ float tile[22][24];
  __shared__ float kk[49];
  int tx = threadIdx.x, ty = threadIdx.y;
  int t  = ty * 16 + tx;
  if (t < 49) kk[t] = kern[b * 64 + t];
  int x0 = blockIdx.x * 16 - 3, y0 = blockIdx.y * 16 - 3;
  const float* src = in + (size_t)img * HWNUM;
  for (int i = t; i < 22 * 22; i += 256) {
    int r = i / 22, c = i % 22;
    int gy = y0 + r, gx = x0 + c;
    float v = 0.f;
    if (gy >= 0 && gy < HH && gx >= 0 && gx < HH) v = src[gy * HH + gx];
    tile[r][c] = v;
  }
  __syncthreads();
  float acc = 0.f;
#pragma unroll
  for (int dy = 0; dy < 7; ++dy)
#pragma unroll
    for (int dx = 0; dx < 7; ++dx)
      acc = fmaf(tile[ty + dy][tx + dx], kk[dy * 7 + dx], acc);
  int oy = blockIdx.y * 16 + ty, ox = blockIdx.x * 16 + tx;
  outp[(size_t)img * HWNUM + oy * HH + ox] = f2bf(acc);
}

// ------------------------- wr -> bf16 conversion ---------------------------

__global__ void afe_cvt_wr(const float* __restrict__ wr, unsigned short* __restrict__ wb) {
  int i = blockIdx.x * blockDim.x + threadIdx.x;
  if (i < 65536) wb[i] = f2bf(wr[i]);
}

__global__ void afe_zero(float* p, int n) {
  int i = blockIdx.x * blockDim.x + threadIdx.x;
  if (i < n) p[i] = 0.f;
}

// -------------------- channel mix GEMM via v_wmma bf16 ---------------------
// D[o,p] = sum_c W[o,c] * X[c,p], per batch.  One 512-thread block = 16 waves
// covers all 256 output channels of a 64-pixel column; each wave holds 4
// accumulator tiles sharing one A fragment per K-step (4 WMMAs / K-step).
// X already bf16 from the conv; staged transposed through LDS once.

__global__ void __launch_bounds__(512)
afe_chanmix(const unsigned short* __restrict__ X, const unsigned short* __restrict__ Wb,
            float* __restrict__ D) {
  __shared__ __align__(16) unsigned short Xs[64][XSTR];   // [pixel][c-in-step]
  int b  = blockIdx.z;
  int p0 = blockIdx.x * 64;
  const unsigned short* Xb = X + (size_t)b * CCH * HWNUM;
  float*                Db = D + (size_t)b * CCH * HWNUM;
  int t    = threadIdx.x;
  int lane = t & 31;
  int wv   = t >> 5;               // 0..15 -> o-tile
  int n    = lane & 15;            // D column (pixel) / A row (o)
  int half = lane >> 4;
  int o0   = wv * 16;
  // cooperative loader coordinates: 32 channels x 64 pixels, 4 bf16 per thread
  int lc = t >> 4;                 // 0..31 channel-in-step
  int lp = (t & 15) * 4;           // 0,4,...,60 pixel

  v8f acc0 = {}, acc1 = {}, acc2 = {}, acc3 = {};
  for (int k0 = 0; k0 < 256; k0 += 32) {
    {
      union { uint2 u; unsigned short s[4]; } ld;
      ld.u = *(const uint2*)(Xb + (size_t)(k0 + lc) * HWNUM + p0 + lp);
      Xs[lp + 0][lc] = ld.s[0];
      Xs[lp + 1][lc] = ld.s[1];
      Xs[lp + 2][lc] = ld.s[2];
      Xs[lp + 3][lc] = ld.s[3];
    }
    __syncthreads();

    union { v16bf v; uint4 q[2]; } a, b0, b1, b2, b3;
    // A (16x32 bf16): lane row m=n; element j -> K = 16*half + j
    const unsigned short* wrow = Wb + (o0 + n) * 256 + k0;
    a.q[0] = *(const uint4*)(wrow + 16 * half);
    a.q[1] = *(const uint4*)(wrow + 16 * half + 8);
    // B (32x16 bf16): lane col n; element j -> K = 16*half + j
    const uint4* x0q = (const uint4*)(&Xs[n][half * 16]);
    const uint4* x1q = (const uint4*)(&Xs[16 + n][half * 16]);
    const uint4* x2q = (const uint4*)(&Xs[32 + n][half * 16]);
    const uint4* x3q = (const uint4*)(&Xs[48 + n][half * 16]);
    b0.q[0] = x0q[0]; b0.q[1] = x0q[1];
    b1.q[0] = x1q[0]; b1.q[1] = x1q[1];
    b2.q[0] = x2q[0]; b2.q[1] = x2q[1];
    b3.q[0] = x3q[0]; b3.q[1] = x3q[1];

    acc0 = __builtin_amdgcn_wmma_f32_16x16x32_bf16(false, a.v, false, b0.v,
                                                   (short)0, acc0, false, false);
    acc1 = __builtin_amdgcn_wmma_f32_16x16x32_bf16(false, a.v, false, b1.v,
                                                   (short)0, acc1, false, false);
    acc2 = __builtin_amdgcn_wmma_f32_16x16x32_bf16(false, a.v, false, b2.v,
                                                   (short)0, acc2, false, false);
    acc3 = __builtin_amdgcn_wmma_f32_16x16x32_bf16(false, a.v, false, b3.v,
                                                   (short)0, acc3, false, false);
    __syncthreads();
  }
#pragma unroll
  for (int r = 0; r < 8; ++r) {
    int o = o0 + r + 8 * half;
    float* orow = Db + (size_t)o * HWNUM + p0 + n;
    orow[0]  = acc0[r];
    orow[16] = acc1[r];
    orow[32] = acc2[r];
    orow[48] = acc3[r];
  }
}

// --------------------- inverse FFT (+ifftshift) + residual -----------------
// ifft2(ifftshift(Y)) == (-1)^(r+c) * ifft2(Y) for even N.

__global__ void afe_ifft2_add(const float* __restrict__ rrp, const float* __restrict__ iip,
                              const float* __restrict__ x, float* __restrict__ out) {
  extern __shared__ float lds[];
  float* re = lds;
  float* im = lds + HH * LSTR;
  int img = blockIdx.x;
  const float* sr = rrp + (size_t)img * HWNUM;
  const float* si = iip + (size_t)img * HWNUM;
  for (int i = threadIdx.x; i < HWNUM; i += blockDim.x) {
    int r = i >> 7, c = i & 127;
    int bc = bitrev7(c);
    re[r * LSTR + bc] = sr[i];
    im[r * LSTR + bc] = si[i];
  }
  __syncthreads();
  afe_fft_rows(re, im, +1.f);
  afe_swap_rows_bitrev(re, im);
  afe_fft_cols(re, im, +1.f);
  const float* xs = x + (size_t)img * HWNUM;
  float* dst = out + (size_t)img * HWNUM;
  for (int i = threadIdx.x; i < HWNUM; i += blockDim.x) {
    int r = i >> 7, c = i & 127;
    float v = re[r * LSTR + c] * 0.0078125f;   // 1/128 ortho
    if ((r + c) & 1) v = -v;
    dst[i] = v + xs[i];
  }
}

// ------------------------------- launcher ----------------------------------

extern "C" void kernel_launch(void* const* d_in, const int* in_sizes, int n_in,
                              void* d_out, int out_size, void* d_ws, size_t ws_size,
                              hipStream_t stream) {
  (void)in_sizes; (void)n_in; (void)out_size; (void)ws_size;
  const float* x  = (const float*)d_in[0];
  const float* w1 = (const float*)d_in[1];
  const float* b1 = (const float*)d_in[2];
  const float* w2 = (const float*)d_in[3];
  const float* b2 = (const float*)d_in[4];
  const float* wr = (const float*)d_in[5];
  float* out = (float*)d_out;

  const size_t N = (size_t)8 * CCH * HWNUM;   // 33,554,432
  float* f   = (float*)d_ws;
  float* xfr = f;                              // reused as rr after conv
  float* xfi = f + N;                          // reused as ii after conv
  unsigned short* cfr16 = (unsigned short*)(f + 2 * N);      // N bf16
  unsigned short* cfi16 = cfr16 + N;                         // N bf16
  float* mag49 = f + 3 * N;                    // 8*64
  float* kern  = mag49 + 8 * 64;               // 8*64
  unsigned short* wb = (unsigned short*)(kern + 8 * 64);     // 256*256 bf16

  size_t ldsBytes = (size_t)2 * HH * LSTR * sizeof(float);   // ~130 KB (<320 KB/WGP)

  afe_zero<<<2, 256, 0, stream>>>(mag49, 8 * 64);
  afe_fft2_fwd<<<2048, 256, ldsBytes, stream>>>(x, xfr, xfi, mag49);
  afe_mlp_kern<<<8, 64, 0, stream>>>(mag49, w1, b1, w2, b2, kern);
  afe_cvt_wr<<<256, 256, 0, stream>>>(wr, wb);

  dim3 cb(16, 16);
  afe_conv7<<<dim3(8, 8, 2048), cb, 0, stream>>>(xfr, cfr16, kern);
  afe_conv7<<<dim3(8, 8, 2048), cb, 0, stream>>>(xfi, cfi16, kern);

  afe_chanmix<<<dim3(256, 1, 8), 512, 0, stream>>>(cfr16, wb, xfr);
  afe_chanmix<<<dim3(256, 1, 8), 512, 0, stream>>>(cfi16, wb, xfi);

  afe_ifft2_add<<<2048, 256, ldsBytes, stream>>>(xfr, xfi, x, out);
}
